// BFNet_59871844106323
// MI455X (gfx1250) — compile-verified, matrix operand order checked by reading, not run on hardware
//
#include <hip/hip_runtime.h>
#include <cstddef>
#include <cstdint>

typedef __attribute__((ext_vector_type(16))) _Float16 v16h;
typedef __attribute__((ext_vector_type(8)))  float    v8f;
typedef __attribute__((ext_vector_type(4)))  uint32_t u32x4;
typedef __attribute__((ext_vector_type(8)))  uint32_t u32x8;

#define N_ELEC 16
#define N_UP 8
#define N_NUC 4
#define NBR 20            // 16 electrons + 4 nuclei
#define BASIS 32
#define KDIM 64
#define EDIM 128
#define N_INTER 3
#define LOG2F_   0.6931471805599453f
#define LOG2E_   1.4426950408889634f

// raw hardware transcendentals (v_exp_f32 / v_log_f32, no guard code)
__device__ __forceinline__ float fast_exp(float x) {   // e^x
    return __builtin_amdgcn_exp2f(x * LOG2E_);
}

// shifted softplus, branch-free, log2-space:
// ssp(x) = max(x,0) - log2 + ln2 * log2(1 + exp2(-|x|*log2e))
// log argument is in [1,2] -> raw v_log_f32 is safe.
__device__ __forceinline__ float sspf(float v) {
    float t = __builtin_amdgcn_exp2f(fabsf(v) * -LOG2E_);
    float l = __builtin_amdgcn_logf(1.0f + t);
    return fmaf(LOG2F_, l, fmaxf(v, 0.0f) - LOG2F_);
}

// K index of element e (0..15) of a 16-bit A/B fragment for lane-half `hi`
__device__ __forceinline__ int frag_k(int e, int hi) {
    return ((e >> 3) << 4) + (hi << 3) + (e & 7);
}

// A fragment (16x32 slice at column k0) from LDS f16 row-major [16][ld]
__device__ __forceinline__ v16h load_a_lds(const _Float16* base, int k0, int ld) {
    int l = threadIdx.x & 31, m = l & 15, hi = l >> 4;
    v16h a;
#pragma unroll
    for (int e = 0; e < 16; ++e) a[e] = base[m * ld + k0 + frag_k(e, hi)];
    return a;
}

// A fragment from x16 restricted to one spin block (8 electron rows, duplicated)
__device__ __forceinline__ v16h load_a_x16_spin(const _Float16* base, int spin, int k0) {
    int l = threadIdx.x & 31, m = l & 15, hi = l >> 4;
    int row = spin * 8 + (m & 7);
    v16h a;
#pragma unroll
    for (int e = 0; e < 16; ++e) a[e] = base[row * EDIM + k0 + frag_k(e, hi)];
    return a;
}

// B fragment (32x16 slice at (k0, col0)) gathered from global f32 row-major [K][ldn]
__device__ __forceinline__ v16h load_b_gf32(const float* __restrict__ base, int k0,
                                            int col0, int ldn) {
    int l = threadIdx.x & 31, n = l & 15, hi = l >> 4;
    v16h b;
#pragma unroll
    for (int e = 0; e < 16; ++e) {
        int k = k0 + frag_k(e, hi);
        b[e] = (_Float16)base[(size_t)k * ldn + (col0 + n)];
    }
    return b;
}

// B fragment from Slater W0[o] (128 x 25 row-major), zero-padded to 32 cols
__device__ __forceinline__ v16h load_b_w0(const float* __restrict__ base, int k0,
                                          int col0) {
    int l = threadIdx.x & 31, n = l & 15, hi = l >> 4;
    int col = col0 + n;
    v16h b;
#pragma unroll
    for (int e = 0; e < 16; ++e) {
        int k = k0 + frag_k(e, hi);
        b[e] = (col < 25) ? (_Float16)base[k * 25 + col] : (_Float16)0.0f;
    }
    return b;
}

// B fragment from LDS-resident f32 matrix (TDM-staged weights)
__device__ __forceinline__ v16h load_b_ldsf32(const float* base, int k0,
                                              int col0, int ldn) {
    int l = threadIdx.x & 31, n = l & 15, hi = l >> 4;
    v16h b;
#pragma unroll
    for (int e = 0; e < 16; ++e) {
        int k = k0 + frag_k(e, hi);
        b[e] = (_Float16)base[k * ldn + (col0 + n)];
    }
    return b;
}

// C accumulator initialized with a per-column bias broadcast over rows
__device__ __forceinline__ v8f bias_frag(const float* __restrict__ bias, int col0) {
    int n = threadIdx.x & 15;
    float bv = bias[col0 + n];
    v8f c;
#pragma unroll
    for (int r = 0; r < 8; ++r) c[r] = bv;
    return c;
}

__device__ __forceinline__ v8f wmma16(v16h a, v16h b, v8f c) {
    return __builtin_amdgcn_wmma_f32_16x16x32_f16(false, a, false, b, (short)0, c,
                                                  false, false);
}

// Gaussian basis A-fragment: e[row, k] = exp(-((d_row - mu_k)^2 / sig_k^2))
__device__ __forceinline__ v16h make_e_frag(const float* s_dists, int mt) {
    int l = threadIdx.x & 31, m = l & 15, hi = l >> 4;
    float d = s_dists[mt * 16 + m];
    v16h a;
#pragma unroll
    for (int e = 0; e < 16; ++e) {
        int k = frag_k(e, hi);
        float q   = (float)k * (1.0f / 31.0f);
        float mu  = 10.0f * q * q;
        float sig = (1.0f + 10.0f * q) * (1.0f / 7.0f);
        float t   = (d - mu) / sig;
        a[e] = (_Float16)__builtin_amdgcn_exp2f(t * t * -LOG2E_);
    }
    return a;
}

// LDS byte offset of a shared-memory object (flat addr low 32 bits == LDS offset)
template <typename T>
__device__ __forceinline__ uint32_t lds_off(const T* p) {
    return (uint32_t)(size_t)p;
}

// Issue TENSOR_LOAD_TO_LDS of `n_elem` f32 (1-row tile) from global -> LDS.
// D# built per ISA 8.3/8.4; VADDR2/3 omitted (<=2D tensor). Uniform operands.
__device__ __forceinline__ void tdm_load_f32(const float* gsrc, uint32_t lds_byte,
                                             uint32_t n_elem) {
    uint64_t ga = (uint64_t)(size_t)gsrc;
    u32x4 g0;
    g0[0] = 1u;                                   // count=1, load, user mode
    g0[1] = lds_byte;                             // lds_addr
    g0[2] = (uint32_t)(ga & 0xFFFFFFFFu);         // global_addr[31:0]
    g0[3] = (uint32_t)((ga >> 32) & 0x1FFFFFFu)   // global_addr[56:32]
            | (2u << 30);                         // type = 2 (image)
    u32x8 g1;
    g1[0] = 2u << 16;                             // data_size = 4 bytes
    g1[1] = (n_elem & 0xFFFFu) << 16;             // tensor_dim0[15:0]
    g1[2] = (n_elem >> 16) & 0xFFFFu;             // tensor_dim0[31:16]
    g1[3] = (n_elem & 0xFFFFu) << 16;             // tile_dim0
    g1[4] = 0u;                                   // tile_dim1/2 unused
    g1[5] = n_elem;                               // tensor_dim0_stride[31:0]
    g1[6] = 0u;
    g1[7] = 0u;
    asm volatile("tensor_load_to_lds %0, %1" : : "s"(g0), "s"(g1) : "memory");
}

struct Smem {
    float    dists[N_ELEC * NBR];       // 320
    float    x[N_ELEC * EDIM];          // running embedding (f32)
    _Float16 x16[N_ELEC * EDIM];        // f16 mirror for WMMA A
    float    hmsg[N_ELEC * KDIM];
    float    z[N_ELEC * KDIM];          // atomic accumulation target
    _Float16 z16[N_ELEC * KDIM];
    _Float16 uT[8][16 * KDIM];          // per-wave stage-1 tile
    __align__(16) float Y[N_NUC * KDIM];
    __align__(16) float rs[N_ELEC * 3];
    float    coords[N_NUC * 3];
    union {                             // TDM weight stage overlaid with Slater scratch
        __align__(16) float w2s[KDIM * KDIM];   // 16 KB f32, live during interactions
        struct {
            float a1[16 * 8 * 25];      // slater layer-0 activations
            float a2[16 * 8 * 5];       // slater layer-1 activations
        } sl;
    } u;
    float    mat[2][64];                // 8x8 slater matrices
    float    det[2];
    float    esum[N_ELEC];
};

__global__ __launch_bounds__(256)
void BFNet_59871844106323_kernel(
    const float* __restrict__ rs,      const float* __restrict__ coords,
    const float* __restrict__ charges, const float* __restrict__ ion_pot,
    const float* __restrict__ X,       const float* __restrict__ Yg,
    const float* __restrict__ sw1,     const float* __restrict__ swb1,
    const float* __restrict__ sw2,     const float* __restrict__ swb2,
    const float* __restrict__ sh,      const float* __restrict__ sg,
    const float* __restrict__ sgb,
    const float* __restrict__ oW0,     const float* __restrict__ ob0,
    const float* __restrict__ oW1,     const float* __restrict__ ob1,
    const float* __restrict__ oW2,     const float* __restrict__ ob2,
    float* __restrict__ out)
{
    __shared__ Smem s;
    const int tid  = threadIdx.x;
    const int wave = tid >> 5;
    const int lane = tid & 31;
    const int ln   = lane & 15;
    const int lhi  = lane >> 4;
    const int b    = blockIdx.x;

    // ---- init: geometry via async global->LDS copies, embeddings via VALU ----
    const float* rsb = rs + (size_t)b * N_ELEC * 3;
    if (tid < 12) {  // 12 lanes x 16B = 192B = rs for this batch element
        uint32_t dst = lds_off(s.rs) + tid * 16;
        uint32_t off = tid * 16;
        asm volatile("global_load_async_to_lds_b128 %0, %1, %2"
                     : : "v"(dst), "v"(off), "s"(rsb) : "memory");
    }
    if (tid >= 32 && tid < 96) {  // waves 1-2 (full EXEC): 64 x 16B = 1 KB = Y
        uint32_t l = tid - 32;
        uint32_t dst = lds_off(s.Y) + l * 16;
        uint32_t off = l * 16;
        asm volatile("global_load_async_to_lds_b128 %0, %1, %2"
                     : : "v"(dst), "v"(off), "s"(Yg) : "memory");
    }
    for (int t = tid; t < N_NUC * 3; t += 256) s.coords[t] = coords[t];
    for (int t = tid; t < N_ELEC * EDIM; t += 256) {
        int row = t >> 7, d = t & 127;
        float v = X[(row >= N_UP ? EDIM : 0) + d];
        s.x[t] = v;
        s.x16[t] = (_Float16)v;
    }
    asm volatile("s_wait_asynccnt 0" ::: "memory");
    __syncthreads();
    for (int t = tid; t < N_ELEC * NBR; t += 256) {
        int i = t / NBR, j = t % NBR;
        const float* pa = &s.rs[i * 3];
        const float* pb = (j < N_ELEC) ? &s.rs[j * 3] : &s.coords[(j - N_ELEC) * 3];
        float dx = pa[0] - pb[0], dy = pa[1] - pb[1], dz = pa[2] - pb[2];
        float d2 = dx * dx + dy * dy + dz * dz;
        s.dists[t] = (d2 > 0.0f) ? __builtin_amdgcn_sqrtf(d2) : 0.0f;
    }
    __syncthreads();

    _Float16* uT = s.uT[wave];

    // Gaussian-basis A fragments are interaction-invariant: compute once per wave,
    // keep in registers (<=3 m-tiles per wave; wave-uniform trip counts).
    v16h ef[3];
#pragma unroll
    for (int t = 0; t < 3; ++t) {
        int mt = wave + 8 * t;
        if (mt < 20) ef[t] = make_e_frag(s.dists, mt);
    }

    // ---- SchNet interactions ----
    for (int it = 0; it < N_INTER; ++it) {
        const float* w1b = sw1 + it * BASIS * KDIM;
        const float* w2b = sw2 + it * KDIM * KDIM;
        const float* hb  = sh  + it * EDIM * KDIM;
        const float* gb_ = sg  + it * KDIM * EDIM;

        // TDM: DMA this interaction's w2 (64x64 f32, L2-resident) into LDS while
        // the waves run the x@h WMMAs. Tracked on TENSORcnt.
        if (wave == 0) tdm_load_f32(w2b, lds_off(s.u.w2s), KDIM * KDIM);

        // hmsg = x @ h[it]   (waves 0..3, one 16-col tile each); waves 4..7 zero z
        if (wave < 4) {
            int nt = wave;
            v8f c = {};
#pragma unroll
            for (int kc = 0; kc < 4; ++kc) {
                v16h a  = load_a_lds(s.x16, kc * 32, EDIM);
                v16h bb = load_b_gf32(hb, kc * 32, nt * 16, KDIM);
                c = wmma16(a, bb, c);
            }
#pragma unroll
            for (int r = 0; r < 8; ++r)
                s.hmsg[(r + 8 * lhi) * KDIM + nt * 16 + ln] = c[r];
        } else {
            for (int t = tid - 128; t < N_ELEC * KDIM; t += 128) s.z[t] = 0.0f;
        }
        if (wave == 0) __builtin_amdgcn_s_wait_tensorcnt(0);
        __syncthreads();

        // w = ssp(e@w1 + wb1)@w2 + wb2, fused masked aggregation into z via LDS atomics
#pragma unroll
        for (int t = 0; t < 3; ++t) {
            int mt = wave + 8 * t;
            if (mt >= 20) continue;
            v16h ae = ef[t];
#pragma unroll
            for (int nt = 0; nt < 4; ++nt) {
                v8f c = bias_frag(swb1 + it * KDIM, nt * 16);
                v16h bb = load_b_gf32(w1b, 0, nt * 16, KDIM);
                c = wmma16(ae, bb, c);
#pragma unroll
                for (int r = 0; r < 8; ++r)
                    uT[(r + 8 * lhi) * KDIM + nt * 16 + ln] = (_Float16)sspf(c[r]);
            }
            // stage-1 stores and stage-2 A-loads are same-wave LDS traffic
            asm volatile("s_wait_dscnt 0" ::: "memory");
#pragma unroll
            for (int nt = 0; nt < 4; ++nt) {
                v8f c = bias_frag(swb2 + it * KDIM, nt * 16);
                v16h a0 = load_a_lds(uT, 0, KDIM);
                v16h b0 = load_b_ldsf32(s.u.w2s, 0, nt * 16, KDIM);
                c = wmma16(a0, b0, c);
                v16h a1 = load_a_lds(uT, 32, KDIM);
                v16h b1 = load_b_ldsf32(s.u.w2s, 32, nt * 16, KDIM);
                c = wmma16(a1, b1, c);
                int col = nt * 16 + ln;
#pragma unroll
                for (int r = 0; r < 8; ++r) {
                    int row = mt * 16 + r + 8 * lhi;   // (i, j) pair index
                    int i = row / NBR, j = row % NBR;
                    float coef = (j < N_ELEC)
                                     ? ((j == i) ? 0.0f : s.hmsg[j * KDIM + col])
                                     : s.Y[(j - N_ELEC) * KDIM + col];
                    atomicAdd(&s.z[i * KDIM + col], c[r] * coef);
                }
            }
        }
        __syncthreads();
        for (int t = tid; t < N_ELEC * KDIM; t += 256) s.z16[t] = (_Float16)s.z[t];
        __syncthreads();

        // x += ssp(z @ g[it] + gb)   (8 waves, one 16-col tile each over 128 cols)
        {
            int nt = wave;
            v8f c = {};
            v16h a0 = load_a_lds(s.z16, 0, KDIM);
            v16h b0 = load_b_gf32(gb_, 0, nt * 16, EDIM);
            c = wmma16(a0, b0, c);
            v16h a1 = load_a_lds(s.z16, 32, KDIM);
            v16h b1 = load_b_gf32(gb_, 32, nt * 16, EDIM);
            c = wmma16(a1, b1, c);
            int col = nt * 16 + ln;
#pragma unroll
            for (int r = 0; r < 8; ++r) {
                int row = r + 8 * lhi;
                float xv = s.x[row * EDIM + col] + sspf(c[r] + sgb[it * EDIM + col]);
                s.x[row * EDIM + col]   = xv;
                s.x16[row * EDIM + col] = (_Float16)xv;
            }
        }
        __syncthreads();
    }

    // ---- Slater layer 0 via WMMA: per orbital o, [8elec x 128] @ W0[o] (128x25) ----
    float* a1 = s.u.sl.a1;
    float* a2 = s.u.sl.a2;
    for (int task = wave; task < 32; task += 8) {   // 16 orbitals x 2 col-tiles
        int og = task >> 1, nt = task & 1;
        int spin = og >> 3;
        const float* w0p = oW0 + (size_t)og * EDIM * 25;
        v8f c = {};
#pragma unroll
        for (int kc = 0; kc < 4; ++kc) {
            v16h a  = load_a_x16_spin(s.x16, spin, kc * 32);
            v16h bb = load_b_w0(w0p, kc * 32, nt * 16);
            c = wmma16(a, bb, c);
        }
        int col = nt * 16 + ln;
        if (lhi == 0 && col < 25) {   // rows 8..15 are duplicates; cols >=25 padding
#pragma unroll
            for (int r = 0; r < 8; ++r)
                a1[(og * 8 + r) * 25 + col] = sspf(c[r] + ob0[og * 25 + col]);
        }
    }
    __syncthreads();

    // ---- Slater layers 1..2 (25 -> 5 -> 1), small VALU tail ----
    for (int idx = tid; idx < 16 * 8 * 5; idx += 256) {
        int g2 = idx % 5, t2 = idx / 5;
        int n = t2 % 8, og = t2 / 8;
        float acc = ob1[og * 5 + g2];
        const float* w1p = oW1 + og * 25 * 5 + g2;
        const float* ap  = &a1[(og * 8 + n) * 25];
        for (int h = 0; h < 25; ++h) acc += ap[h] * w1p[h * 5];
        a2[idx] = sspf(acc);
    }
    __syncthreads();
    for (int idx = tid; idx < 16 * 8; idx += 256) {
        int n = idx % 8, og = idx / 8;
        float acc = ob2[og];
        const float* w2p = oW2 + og * 5;
        const float* ap  = &a2[(og * 8 + n) * 5];
        for (int g2 = 0; g2 < 5; ++g2) acc += ap[g2] * w2p[g2];
        s.mat[og >> 3][n * 8 + (og & 7)] = acc;
    }
    __syncthreads();

    // ---- determinants (LU w/ partial pivoting) + asymptotic factor ----
    if (tid < 2) {
        float* M = s.mat[tid];
        float det = 1.0f;
        for (int k = 0; k < 8; ++k) {
            int p = k;
            float mx = fabsf(M[k * 8 + k]);
            for (int r2 = k + 1; r2 < 8; ++r2) {
                float v = fabsf(M[r2 * 8 + k]);
                if (v > mx) { mx = v; p = r2; }
            }
            if (p != k) {
                for (int c2 = k; c2 < 8; ++c2) {
                    float t = M[k * 8 + c2]; M[k * 8 + c2] = M[p * 8 + c2]; M[p * 8 + c2] = t;
                }
                det = -det;
            }
            float piv = M[k * 8 + k];
            det *= piv;
            float inv = 1.0f / piv;
            for (int r2 = k + 1; r2 < 8; ++r2) {
                float f = M[r2 * 8 + k] * inv;
                for (int c2 = k + 1; c2 < 8; ++c2) M[r2 * 8 + c2] -= f * M[k * 8 + c2];
            }
        }
        s.det[tid] = det;
    }
    if (tid >= 32 && tid < 48) {
        int i = tid - 32;
        float decay = __builtin_amdgcn_sqrtf(2.0f * ion_pot[0]);
        float sum = 0.0f;
        for (int m = 0; m < N_NUC; ++m) {
            float d = s.dists[i * NBR + N_ELEC + m];
            sum += fast_exp(-(charges[m] * d + decay * d * d) / (1.0f + d));
        }
        s.esum[i] = sum;
    }
    __syncthreads();
    if (tid == 0) {
        float p = s.det[0] * s.det[1];
        for (int i = 0; i < N_ELEC; ++i) p *= s.esum[i];
        out[b] = p;
    }
}

extern "C" void kernel_launch(void* const* d_in, const int* in_sizes, int n_in,
                              void* d_out, int out_size, void* d_ws, size_t ws_size,
                              hipStream_t stream) {
    (void)n_in; (void)d_ws; (void)ws_size; (void)out_size;
    const float* rs      = (const float*)d_in[0];
    const float* coords  = (const float*)d_in[1];
    const float* charges = (const float*)d_in[2];
    const float* ion_pot = (const float*)d_in[3];
    const float* X       = (const float*)d_in[4];
    const float* Yg      = (const float*)d_in[5];
    const float* sw1     = (const float*)d_in[6];
    const float* swb1    = (const float*)d_in[7];
    const float* sw2     = (const float*)d_in[8];
    const float* swb2    = (const float*)d_in[9];
    const float* sh      = (const float*)d_in[10];
    const float* sg      = (const float*)d_in[11];
    const float* sgb     = (const float*)d_in[12];
    const float* oW0     = (const float*)d_in[13];
    const float* ob0     = (const float*)d_in[14];
    const float* oW1     = (const float*)d_in[15];
    const float* ob1     = (const float*)d_in[16];
    const float* oW2     = (const float*)d_in[17];
    const float* ob2     = (const float*)d_in[18];
    float* out = (float*)d_out;

    int B = in_sizes[0] / (N_ELEC * 3);
    BFNet_59871844106323_kernel<<<B, 256, 0, stream>>>(
        rs, coords, charges, ion_pot, X, Yg,
        sw1, swb1, sw2, swb2, sh, sg, sgb,
        oW0, ob0, oW1, ob1, oW2, ob2, out);
}
